// DIT_Syn_Cls_Pep_Cls_AMP_4655744549575
// MI455X (gfx1250) — compile-verified, matrix operand order checked by reading, not run on hardware
//
#include <hip/hip_runtime.h>
#include <hip/hip_bf16.h>

typedef __attribute__((ext_vector_type(16))) _Float16 v16h;
typedef __attribute__((ext_vector_type(8)))  float    v8f;

// ---------------------------------------------------------------------------
// WMMA helpers (gfx1250, wave32)
// ---------------------------------------------------------------------------
__device__ __forceinline__ v8f v8f_zero() {
    v8f z;
#pragma unroll
    for (int e = 0; e < 8; ++e) z[e] = 0.0f;
    return z;
}

__device__ __forceinline__ v8f wmma32(v16h a, v16h b, v8f c) {
    return __builtin_amdgcn_wmma_f32_16x16x32_f16(
        /*neg_a=*/false, a, /*neg_b=*/false, b,
        /*c_mod=*/(short)0, c, /*reuse_a=*/false, /*reuse_b=*/false);
}

// Load a 16x32 f16 fragment (A-layout; also used for B when the matrix is
// supplied as BT, i.e. N-major with K contiguous).  ISA 16-bit A layout:
// lanes 0-15 hold row m = lane, K = 0..7 (v0..v3) and 16..23 (v4..v7);
// lanes 16-31 hold K = 8..15 and 24..31.
__device__ __forceinline__ v16h frag_ld(const _Float16* p, int ld, int lane) {
    const int hf = (lane >> 4) & 1;
    const int r  = lane & 15;
    v16h f;
#pragma unroll
    for (int i = 0; i < 8; ++i) {
        const int kb = ((i >> 2) << 4) + (hf << 3) + ((i & 3) << 1);
        f[2 * i]     = p[r * ld + kb];
        f[2 * i + 1] = p[r * ld + kb + 1];
    }
    return f;
}

// Async global->LDS copy of 16 bytes per lane (gfx1250 ASYNCcnt path).
// lds_off: LDS byte offset (low 32 bits of the generic shared address);
// goff: per-lane byte offset; base: uniform 64-bit base in SGPRs.
__device__ __forceinline__ void async_copy_b128(unsigned lds_off, unsigned goff,
                                                const void* base) {
    asm volatile("global_load_async_to_lds_b128 %0, %1, %2"
                 :: "v"(lds_off), "v"(goff), "s"(base)
                 : "memory");
}

__device__ __forceinline__ void wait_async0() {
    asm volatile("s_wait_asynccnt 0x0" ::: "memory");
}

// ---------------------------------------------------------------------------
// Weight convert + transpose: W (K x N, f32) -> WT (N x K, f16)
// ---------------------------------------------------------------------------
__global__ __launch_bounds__(256) void convT_kernel(const float* __restrict__ W,
                                                    _Float16* __restrict__ WT,
                                                    int K, int N) {
    __shared__ float tile[32][33];
    const int tx = threadIdx.x & 31;
    const int ty = threadIdx.x >> 5;     // 0..7
    const int n0 = blockIdx.x * 32;
    const int k0 = blockIdx.y * 32;
#pragma unroll
    for (int rr = 0; rr < 4; ++rr)
        tile[ty + rr * 8][tx] = W[(size_t)(k0 + ty + rr * 8) * N + n0 + tx];
    __syncthreads();
#pragma unroll
    for (int rr = 0; rr < 4; ++rr)
        WT[(size_t)(n0 + ty + rr * 8) * K + k0 + tx] = (_Float16)tile[tx][ty + rr * 8];
}

// ---------------------------------------------------------------------------
// mod = c @ adaLN_w + adaLN_b   (4 x 6144)
// ---------------------------------------------------------------------------
__global__ __launch_bounds__(256) void cond_mod_kernel(const float* __restrict__ c,
                                                       const float* __restrict__ W,
                                                       const float* __restrict__ bias,
                                                       float* __restrict__ mod) {
    __shared__ float cb[128];
    const int b = blockIdx.y;
    const int n = blockIdx.x * 256 + threadIdx.x;
    if (threadIdx.x < 128) cb[threadIdx.x] = c[b * 128 + threadIdx.x];
    __syncthreads();
    float acc = bias[n];
#pragma unroll 8
    for (int k = 0; k < 128; ++k) acc += cb[k] * W[(size_t)k * 6144 + n];
    mod[(size_t)b * 6144 + n] = acc;
}

// ---------------------------------------------------------------------------
// LayerNorm + modulate -> f16.  One row (D=1024) per 256-thread block.
// ---------------------------------------------------------------------------
__global__ __launch_bounds__(256) void ln_mod_kernel(const float* __restrict__ x,
                                                     const float* __restrict__ w,
                                                     const float* __restrict__ mod,
                                                     int shift_off, int scale_off,
                                                     _Float16* __restrict__ out) {
    const int r = blockIdx.x;
    const int b = r >> 10;  // S = 1024
    const float* xr = x + (size_t)r * 1024;
    float v[4], s = 0.0f, ss = 0.0f;
#pragma unroll
    for (int u = 0; u < 4; ++u) {
        const int col = threadIdx.x + u * 256;
        v[u] = xr[col];
        s += v[u];
        ss += v[u] * v[u];
    }
#pragma unroll
    for (int off = 16; off >= 1; off >>= 1) {
        s  += __shfl_xor(s, off, 32);
        ss += __shfl_xor(ss, off, 32);
    }
    __shared__ float ws1[8], ws2[8];
    if ((threadIdx.x & 31) == 0) {
        ws1[threadIdx.x >> 5] = s;
        ws2[threadIdx.x >> 5] = ss;
    }
    __syncthreads();
    float ts = 0.0f, tss = 0.0f;
#pragma unroll
    for (int i = 0; i < 8; ++i) { ts += ws1[i]; tss += ws2[i]; }
    const float mu  = ts * (1.0f / 1024.0f);
    const float var = tss * (1.0f / 1024.0f) - mu * mu;
    const float rs  = rsqrtf(var + 1e-5f);
    const float* mrow = mod + (size_t)b * 6144;
#pragma unroll
    for (int u = 0; u < 4; ++u) {
        const int col = threadIdx.x + u * 256;
        const float y = (v[u] - mu) * rs * w[col];
        const float o = y * (1.0f + mrow[scale_off + col]) + mrow[shift_off + col];
        out[(size_t)r * 1024 + col] = (_Float16)o;
    }
}

// ---------------------------------------------------------------------------
// RoPE + split qkv (b,s,3,h,hd) f16 -> Q/K/V in (b*h, s, hd) f16
// ---------------------------------------------------------------------------
__global__ __launch_bounds__(256) void rope_split_kernel(const _Float16* __restrict__ qkv,
                                                         const float* __restrict__ cosb,
                                                         const float* __restrict__ sinb,
                                                         _Float16* __restrict__ qo,
                                                         _Float16* __restrict__ ko,
                                                         _Float16* __restrict__ vo) {
    const int idx = blockIdx.x * 256 + threadIdx.x;   // B*H*S*32 = 2M threads
    const int d = idx & 31;
    const int s = (idx >> 5) & 1023;
    const int h = (idx >> 15) & 15;
    const int b = idx >> 19;
    const size_t base = (size_t)(b * 1024 + s) * 3072 + h * 64 + d;
    const float q1 = (float)qkv[base],        q2 = (float)qkv[base + 32];
    const float k1 = (float)qkv[base + 1024], k2 = (float)qkv[base + 1024 + 32];
    const float v1 = (float)qkv[base + 2048], v2 = (float)qkv[base + 2048 + 32];
    const float cc = cosb[s * 32 + d];
    const float sn = sinb[s * 32 + d];
    const size_t ob = (size_t)((b * 16 + h) * 1024 + s) * 64 + d;
    qo[ob]      = (_Float16)(q1 * cc - q2 * sn);
    qo[ob + 32] = (_Float16)(q2 * cc + q1 * sn);
    ko[ob]      = (_Float16)(k1 * cc - k2 * sn);
    ko[ob + 32] = (_Float16)(k2 * cc + k1 * sn);
    vo[ob]      = (_Float16)v1;
    vo[ob + 32] = (_Float16)v2;
}

// ---------------------------------------------------------------------------
// WMMA GEMM: C[MxN] = A[MxK] (f16) * B, with B given as BT[NxK] (f16).
// Block tile 128x128x32, 8 waves, each wave 32x64 (2x4 16x16 tiles).
// Double-buffered LDS staged with GLOBAL_LOAD_ASYNC_TO_LDS_B128 (ASYNCcnt).
// EPI 0: store f16      (qkv projection)
// EPI 1: +bias, GELU, store f16 (MLP up)
// EPI 2: resid + gate*acc        -> f32 (attn out-proj)
// EPI 3: resid + gate*(acc+bias) -> f32 (MLP down, final output)
// ---------------------------------------------------------------------------
template <int EPI>
__global__ __launch_bounds__(256) void gemm_wmma_kernel(
    const _Float16* __restrict__ A, const _Float16* __restrict__ BT,
    int M, int N, int K,
    float* __restrict__ Cf, _Float16* __restrict__ Ch,
    const float* __restrict__ bias, const float* __restrict__ mod, int gate_off,
    const float* __restrict__ resid) {
    constexpr int BK = 32, LD = BK + 8;  // 80B row stride (16B aligned)
    __shared__ alignas(16) _Float16 As[2][128][LD];
    __shared__ alignas(16) _Float16 Bs[2][128][LD];
    const int t = threadIdx.x, lane = t & 31, wave = t >> 5;
    const int wr = wave >> 1, wc = wave & 1;
    const int m0 = blockIdx.y * 128, n0 = blockIdx.x * 128;

    v8f acc[2][4];
#pragma unroll
    for (int mi = 0; mi < 2; ++mi)
#pragma unroll
        for (int j = 0; j < 4; ++j) acc[mi][j] = v8f_zero();

    // Per-thread staging coordinates (two passes cover 128 rows x 32 cols).
    int rS[2], cS[2];
#pragma unroll
    for (int p = 0; p < 2; ++p) {
        const int lid = p * 256 + t;
        rS[p] = lid >> 2;           // 0..127
        cS[p] = (lid & 3) << 3;     // 0,8,16,24 halves
    }

    auto stage = [&](int buf, int kt) {
        const int k0 = kt * BK;
#pragma unroll
        for (int p = 0; p < 2; ++p) {
            const unsigned ldsA = (unsigned)(size_t)&As[buf][rS[p]][cS[p]];
            const unsigned offA = (unsigned)(((size_t)(m0 + rS[p]) * K + k0 + cS[p]) * 2);
            async_copy_b128(ldsA, offA, A);
            const unsigned ldsB = (unsigned)(size_t)&Bs[buf][rS[p]][cS[p]];
            const unsigned offB = (unsigned)(((size_t)(n0 + rS[p]) * K + k0 + cS[p]) * 2);
            async_copy_b128(ldsB, offB, BT);
        }
    };

    const int kTiles = K / BK;
    stage(0, 0);
    wait_async0();
    __syncthreads();

    for (int kt = 0; kt < kTiles; ++kt) {
        const int buf = kt & 1;
        if (kt + 1 < kTiles) stage(buf ^ 1, kt + 1);   // prefetch next tile async

        const v16h a0 = frag_ld(&As[buf][wr * 32][0],      LD, lane);
        const v16h a1 = frag_ld(&As[buf][wr * 32 + 16][0], LD, lane);
#pragma unroll
        for (int j = 0; j < 4; ++j) {
            const v16h bj = frag_ld(&Bs[buf][wc * 64 + j * 16][0], LD, lane);
            acc[0][j] = wmma32(a0, bj, acc[0][j]);
            acc[1][j] = wmma32(a1, bj, acc[1][j]);
        }
        if (kt + 1 < kTiles) wait_async0();
        __syncthreads();
    }

    const int hf = (lane >> 4) & 1, ln = lane & 15;
#pragma unroll
    for (int mi = 0; mi < 2; ++mi)
#pragma unroll
        for (int j = 0; j < 4; ++j)
#pragma unroll
            for (int i = 0; i < 8; ++i) {
                const int row = m0 + wr * 32 + mi * 16 + hf * 8 + i;
                const int col = n0 + wc * 64 + j * 16 + ln;
                float v = acc[mi][j][i];
                if (EPI == 0) {
                    Ch[(size_t)row * N + col] = (_Float16)v;
                } else if (EPI == 1) {
                    v += bias[col];
                    const float g = 0.5f * v *
                        (1.0f + tanhf(0.7978845608028654f * (v + 0.044715f * v * v * v)));
                    Ch[(size_t)row * N + col] = (_Float16)g;
                } else {
                    if (EPI == 3) v += bias[col];
                    const int b = row >> 10;  // S = 1024
                    const float g = mod[(size_t)b * 6144 + gate_off + col];
                    Cf[(size_t)row * N + col] =
                        resid[(size_t)row * N + col] + g * v;
                }
            }
}

// ---------------------------------------------------------------------------
// Flash attention: one (b,h) + 128 query rows per block; 8 waves, 16 q each.
// Q/K/V in (b*h, s, hd=64) f16.  Out -> attn (b, s, h*64+hd) f16.
// ---------------------------------------------------------------------------
__global__ __launch_bounds__(256) void attn_flash_kernel(
    const _Float16* __restrict__ Q, const _Float16* __restrict__ Kh,
    const _Float16* __restrict__ Vh, _Float16* __restrict__ Oh) {
    constexpr int LD = 72;  // 144B row stride (16B aligned)
    __shared__ alignas(16) _Float16 Qs[128][LD];
    __shared__ alignas(16) _Float16 Ks[64][LD];
    __shared__ alignas(16) _Float16 Vts[64][LD];      // transposed V: [hd][key]
    __shared__ alignas(16) _Float16 Ps[8][16][LD];    // per-wave P scratch
    const int t = threadIdx.x, lane = t & 31, wave = t >> 5;
    const int qb = blockIdx.x * 128;
    const int bh = blockIdx.y;
    const size_t base = (size_t)bh * 1024 * 64;

#pragma unroll
    for (int p = 0; p < 4; ++p) {
        const int lid = p * 256 + t;
        const int r = lid >> 3, c8 = (lid & 7) << 3;
        *reinterpret_cast<float4*>(&Qs[r][c8]) =
            *reinterpret_cast<const float4*>(&Q[base + (size_t)(qb + r) * 64 + c8]);
    }
    __syncthreads();
    const v16h qf0 = frag_ld(&Qs[wave * 16][0],  LD, lane);
    const v16h qf1 = frag_ld(&Qs[wave * 16][32], LD, lane);

    v8f oacc[4];
#pragma unroll
    for (int j = 0; j < 4; ++j) oacc[j] = v8f_zero();
    float rmax[8], rsum[8];
#pragma unroll
    for (int i = 0; i < 8; ++i) { rmax[i] = -1e30f; rsum[i] = 0.0f; }
    const int hf = (lane >> 4) & 1, ln = lane & 15;

    for (int kt = 0; kt < 16; ++kt) {   // 16 key tiles of 64
#pragma unroll
        for (int p = 0; p < 2; ++p) {
            const int lid = p * 256 + t;
            const int r = lid >> 3, c8 = (lid & 7) << 3;
            *reinterpret_cast<float4*>(&Ks[r][c8]) =
                *reinterpret_cast<const float4*>(&Kh[base + (size_t)(kt * 64 + r) * 64 + c8]);
            const float4 vv =
                *reinterpret_cast<const float4*>(&Vh[base + (size_t)(kt * 64 + r) * 64 + c8]);
            const _Float16* vp = reinterpret_cast<const _Float16*>(&vv);
#pragma unroll
            for (int u = 0; u < 8; ++u) Vts[c8 + u][r] = vp[u];
        }
        __syncthreads();

        // S = Q @ K^T : Ks rows (keys x hd) are exactly BT layout.
        v8f sa[4];
#pragma unroll
        for (int j = 0; j < 4; ++j) {
            sa[j] = v8f_zero();
            const v16h b0 = frag_ld(&Ks[j * 16][0],  LD, lane);
            const v16h b1 = frag_ld(&Ks[j * 16][32], LD, lane);
            sa[j] = wmma32(qf0, b0, sa[j]);
            sa[j] = wmma32(qf1, b1, sa[j]);
        }

        // Online softmax (row reductions across the 16-lane group).
#pragma unroll
        for (int i = 0; i < 8; ++i) {
            float pv[4], m_loc = -1e30f;
#pragma unroll
            for (int j = 0; j < 4; ++j) {
                pv[j] = sa[j][i] * 0.125f;   // 1/sqrt(64)
                m_loc = fmaxf(m_loc, pv[j]);
            }
#pragma unroll
            for (int off = 8; off >= 1; off >>= 1)
                m_loc = fmaxf(m_loc, __shfl_xor(m_loc, off, 32));
            const float nm = fmaxf(rmax[i], m_loc);
            const float corr = __expf(rmax[i] - nm);
            rmax[i] = nm;
            float s_loc = 0.0f;
#pragma unroll
            for (int j = 0; j < 4; ++j) {
                pv[j] = __expf(pv[j] - nm);
                s_loc += pv[j];
            }
#pragma unroll
            for (int off = 8; off >= 1; off >>= 1)
                s_loc += __shfl_xor(s_loc, off, 32);
            rsum[i] = rsum[i] * corr + s_loc;
#pragma unroll
            for (int j = 0; j < 4; ++j) {
                oacc[j][i] *= corr;
                Ps[wave][hf * 8 + i][j * 16 + ln] = (_Float16)pv[j];
            }
        }
        // Per-wave LDS round-trip: DS ops are in-order per wave; drain DScnt.
        asm volatile("s_wait_dscnt 0x0" ::: "memory");
        const v16h pa0 = frag_ld(&Ps[wave][0][0],  LD, lane);
        const v16h pa1 = frag_ld(&Ps[wave][0][32], LD, lane);
#pragma unroll
        for (int j = 0; j < 4; ++j) {
            const v16h vb0 = frag_ld(&Vts[j * 16][0],  LD, lane);
            const v16h vb1 = frag_ld(&Vts[j * 16][32], LD, lane);
            oacc[j] = wmma32(pa0, vb0, oacc[j]);
            oacc[j] = wmma32(pa1, vb1, oacc[j]);
        }
        __syncthreads();
    }

    const int b = bh >> 4, h = bh & 15;
#pragma unroll
    for (int j = 0; j < 4; ++j)
#pragma unroll
        for (int i = 0; i < 8; ++i) {
            const int srow = qb + wave * 16 + hf * 8 + i;
            const int col  = h * 64 + j * 16 + ln;
            Oh[(size_t)(b * 1024 + srow) * 1024 + col] =
                (_Float16)(oacc[j][i] / rsum[i]);
        }
}

// ---------------------------------------------------------------------------
// Host-side orchestration
// ---------------------------------------------------------------------------
extern "C" void kernel_launch(void* const* d_in, const int* in_sizes, int n_in,
                              void* d_out, int out_size, void* d_ws, size_t ws_size,
                              hipStream_t stream) {
    (void)in_sizes; (void)n_in; (void)out_size; (void)ws_size;
    const float* x        = (const float*)d_in[0];
    const float* c        = (const float*)d_in[1];
    const float* norm1_w  = (const float*)d_in[2];
    const float* norm2_w  = (const float*)d_in[3];
    const float* w_qkv    = (const float*)d_in[4];
    const float* w_out    = (const float*)d_in[5];
    const float* w1       = (const float*)d_in[6];
    const float* b1       = (const float*)d_in[7];
    const float* w2       = (const float*)d_in[8];
    const float* b2       = (const float*)d_in[9];
    const float* adaLN_w  = (const float*)d_in[10];
    const float* adaLN_b  = (const float*)d_in[11];
    const float* cosb     = (const float*)d_in[12];
    const float* sinb     = (const float*)d_in[13];
    float* out = (float*)d_out;

    // Workspace carve-out (256B aligned)
    char* p = (char*)d_ws;
    auto alloc = [&](size_t bytes) -> void* {
        void* r = (void*)p;
        p += (bytes + 255) & ~(size_t)255;
        return r;
    };
    const size_t BS = 4096;        // B*S rows
    float*    modb   = (float*)alloc(4 * 6144 * sizeof(float));
    _Float16* xm_h   = (_Float16*)alloc(BS * 1024 * 2);
    _Float16* qkv_h  = (_Float16*)alloc(BS * 3072 * 2);
    _Float16* q_h    = (_Float16*)alloc((size_t)64 * 1024 * 64 * 2);
    _Float16* k_h    = (_Float16*)alloc((size_t)64 * 1024 * 64 * 2);
    _Float16* v_h    = (_Float16*)alloc((size_t)64 * 1024 * 64 * 2);
    _Float16* attn_h = (_Float16*)alloc(BS * 1024 * 2);
    float*    x_res  = (float*)alloc(BS * 1024 * sizeof(float));
    _Float16* xm2_h  = (_Float16*)alloc(BS * 1024 * 2);
    _Float16* h_h    = (_Float16*)alloc(BS * 4096 * 2);
    _Float16* wqkvT  = (_Float16*)alloc((size_t)3072 * 1024 * 2);
    _Float16* woutT  = (_Float16*)alloc((size_t)1024 * 1024 * 2);
    _Float16* w1T    = (_Float16*)alloc((size_t)4096 * 1024 * 2);
    _Float16* w2T    = (_Float16*)alloc((size_t)1024 * 4096 * 2);

    // 1) weight convert+transpose
    convT_kernel<<<dim3(3072 / 32, 1024 / 32), 256, 0, stream>>>(w_qkv, wqkvT, 1024, 3072);
    convT_kernel<<<dim3(1024 / 32, 1024 / 32), 256, 0, stream>>>(w_out, woutT, 1024, 1024);
    convT_kernel<<<dim3(4096 / 32, 1024 / 32), 256, 0, stream>>>(w1,    w1T,   1024, 4096);
    convT_kernel<<<dim3(1024 / 32, 4096 / 32), 256, 0, stream>>>(w2,    w2T,   4096, 1024);

    // 2) adaLN modulation vectors
    cond_mod_kernel<<<dim3(6144 / 256, 4), 256, 0, stream>>>(c, adaLN_w, adaLN_b, modb);

    // 3) LN1 + modulate
    ln_mod_kernel<<<4096, 256, 0, stream>>>(x, norm1_w, modb, /*shift*/0, /*scale*/1024, xm_h);

    // 4) QKV projection
    gemm_wmma_kernel<0><<<dim3(3072 / 128, 4096 / 128), 256, 0, stream>>>(
        xm_h, wqkvT, 4096, 3072, 1024, nullptr, qkv_h, nullptr, nullptr, 0, nullptr);

    // 5) RoPE + head split
    rope_split_kernel<<<8192, 256, 0, stream>>>(qkv_h, cosb, sinb, q_h, k_h, v_h);

    // 6) attention
    attn_flash_kernel<<<dim3(1024 / 128, 64), 256, 0, stream>>>(q_h, k_h, v_h, attn_h);

    // 7) out-proj + gate_msa * (.) + residual(x) -> x_res
    gemm_wmma_kernel<2><<<dim3(1024 / 128, 4096 / 128), 256, 0, stream>>>(
        attn_h, woutT, 4096, 1024, 1024, x_res, nullptr, nullptr, modb, /*gate_msa*/2048, x);

    // 8) LN2 + modulate
    ln_mod_kernel<<<4096, 256, 0, stream>>>(x_res, norm2_w, modb, /*shift*/3072, /*scale*/4096, xm2_h);

    // 9) MLP up + bias + GELU
    gemm_wmma_kernel<1><<<dim3(4096 / 128, 4096 / 128), 256, 0, stream>>>(
        xm2_h, w1T, 4096, 4096, 1024, nullptr, h_h, b1, nullptr, 0, nullptr);

    // 10) MLP down + bias + gate_mlp * (.) + residual(x_res) -> out
    gemm_wmma_kernel<3><<<dim3(1024 / 128, 4096 / 128), 256, 0, stream>>>(
        h_h, w2T, 4096, 1024, 4096, out, nullptr, b2, modb, /*gate_mlp*/5120, x_res);
}
